// TokenCentricGraphAttention_25074019074112
// MI455X (gfx1250) — matrix operand order repaired and checked by reading, not compiled, not run on hardware
//
#include <hip/hip_runtime.h>
#include <hip/hip_bf16.h>

// ---------------------------------------------------------------------------
// MI455X (gfx1250) implementation: bf16 WMMA GEMMs + fused attention softmax.
// All matrix math via v_wmma_f32_16x16x32_bf16 (fp32 accumulate).
// ---------------------------------------------------------------------------

typedef __bf16 bf16_t;
typedef __attribute__((ext_vector_type(16))) __bf16 v16bf;
typedef __attribute__((ext_vector_type(8)))  __bf16 v8bf;
typedef __attribute__((ext_vector_type(8)))  float  v8f;

constexpr int Bc = 2, Sc = 512, Pc = 4096, TDc = 512, SDc = 384;
constexpr int Hc = 8, HDc = 64, EDc = 32, FFc = 2048;
constexpr float EPSc = 1e-5f;

__device__ __forceinline__ v8f wmma_bf16(v16bf a, v16bf b, v8f c) {
  // D = A(16x32 bf16) * B(32x16 bf16) + C(16x16 f32)
  return __builtin_amdgcn_wmma_f32_16x16x32_bf16(
      /*neg_a=*/false, a, /*neg_b=*/false, b,
      /*c_mod=*/(short)0, c, /*reuse_a=*/false, /*reuse_b=*/false);
}

// A fragment: row-major source, lane holds row (lane&15); chunks at
// K = half*8 .. +7 and K = 16+half*8 .. +7  (half = lane>>4).
__device__ __forceinline__ v16bf load_a_frag(const bf16_t* __restrict__ base, int ld) {
  const unsigned lane = threadIdx.x & 31u;
  const unsigned row  = lane & 15u;
  const unsigned half = lane >> 4;
  const bf16_t* p = base + (size_t)row * ld + half * 8u;
  v8bf lo = *(const v8bf*)(p);
  v8bf hi = *(const v8bf*)(p + 16);
  return __builtin_shufflevector(lo, hi, 0,1,2,3,4,5,6,7,8,9,10,11,12,13,14,15);
}

// B fragment: column-major source (B^T row-major, stride ld between columns).
// Lane holds column (lane&15); chunks at K = half*8 .. +7 and K = 16+half*8 .. +7.
__device__ __forceinline__ v16bf load_b_frag(const bf16_t* __restrict__ base, int ld) {
  const unsigned lane = threadIdx.x & 31u;
  const unsigned col  = lane & 15u;
  const unsigned half = lane >> 4;
  const bf16_t* p = base + (size_t)col * ld + half * 8u;
  v8bf lo = *(const v8bf*)(p);
  v8bf hi = *(const v8bf*)(p + 16);
  return __builtin_shufflevector(lo, hi, 0,1,2,3,4,5,6,7,8,9,10,11,12,13,14,15);
}

// ---------------------------------------------------------------------------
// Elementwise / reduction kernels
// ---------------------------------------------------------------------------

__global__ __launch_bounds__(128)
void ln_bf16_kernel(const float* __restrict__ x, const float* __restrict__ g,
                    const float* __restrict__ bta, bf16_t* __restrict__ y, int D) {
  const int row = blockIdx.x;
  const float* xr = x + (size_t)row * D;
  float xs[4];
  int cnt = 0;
  float s = 0.f, sq = 0.f;
  for (int idx = threadIdx.x; idx < D; idx += 128) {
    float v = xr[idx]; xs[cnt++] = v; s += v; sq += v * v;
  }
  __shared__ float s1[128], s2[128];
  s1[threadIdx.x] = s; s2[threadIdx.x] = sq;
  __syncthreads();
  for (int off = 64; off > 0; off >>= 1) {
    if ((int)threadIdx.x < off) {
      s1[threadIdx.x] += s1[threadIdx.x + off];
      s2[threadIdx.x] += s2[threadIdx.x + off];
    }
    __syncthreads();
  }
  const float mean = s1[0] / D;
  const float var  = s2[0] / D - mean * mean;
  const float rstd = rsqrtf(var + EPSc);
  cnt = 0;
  for (int idx = threadIdx.x; idx < D; idx += 128) {
    float v = (xs[cnt++] - mean) * rstd * g[idx] + bta[idx];
    y[(size_t)row * D + idx] = (bf16_t)v;
  }
}

__global__ __launch_bounds__(128)
void add_ln_kernel(const float* __restrict__ a, const float* __restrict__ b,
                   const float* __restrict__ g, const float* __restrict__ bta,
                   float* __restrict__ y32, bf16_t* __restrict__ y16, int D) {
  const int row = blockIdx.x;
  const float* ar = a + (size_t)row * D;
  const float* br = b + (size_t)row * D;
  float xs[4];
  int cnt = 0;
  float s = 0.f, sq = 0.f;
  for (int idx = threadIdx.x; idx < D; idx += 128) {
    float v = ar[idx] + br[idx]; xs[cnt++] = v; s += v; sq += v * v;
  }
  __shared__ float s1[128], s2[128];
  s1[threadIdx.x] = s; s2[threadIdx.x] = sq;
  __syncthreads();
  for (int off = 64; off > 0; off >>= 1) {
    if ((int)threadIdx.x < off) {
      s1[threadIdx.x] += s1[threadIdx.x + off];
      s2[threadIdx.x] += s2[threadIdx.x + off];
    }
    __syncthreads();
  }
  const float mean = s1[0] / D;
  const float var  = s2[0] / D - mean * mean;
  const float rstd = rsqrtf(var + EPSc);
  cnt = 0;
  for (int idx = threadIdx.x; idx < D; idx += 128) {
    float v = (xs[cnt++] - mean) * rstd * g[idx] + bta[idx];
    if (y32) y32[(size_t)row * D + idx] = v;
    if (y16) y16[(size_t)row * D + idx] = (bf16_t)v;
  }
}

// W [K,N] fp32 -> Wt [N,K] bf16
__global__ void transpose_bf16_kernel(const float* __restrict__ W, bf16_t* __restrict__ Wt,
                                      int K, int N) {
  size_t idx = (size_t)blockIdx.x * blockDim.x + threadIdx.x;
  if (idx >= (size_t)K * N) return;
  int k = (int)(idx / N), n = (int)(idx % N);
  Wt[(size_t)n * K + k] = (bf16_t)W[idx];
}

// s[h] = dot(edge_emb[h], ep_w)
__global__ void edge_scalar_kernel(const float* __restrict__ ee, const float* __restrict__ w,
                                   float* __restrict__ s_head) {
  int h = threadIdx.x >> 5, lane = threadIdx.x & 31;
  float v = ee[h * EDc + lane] * w[lane];
  for (int off = 16; off > 0; off >>= 1) v += __shfl_xor(v, off, 32);
  if (lane == 0) s_head[h] = v;
}

// ---------------------------------------------------------------------------
// Generic WMMA GEMM: C[M,N] = A[M,K](bf16) @ Wt[N,K](bf16) + bias, epilogues.
// Block = 64x64 tile, 8 wave32s, each wave does a 16x32 strip, K step 32.
// ---------------------------------------------------------------------------
enum { EPI_BF16 = 0, EPI_GELU_BF16 = 1, EPI_F32 = 2, EPI_VT_BF16 = 3 };

template <int EPI>
__global__ __launch_bounds__(256)
void gemm_kernel(const bf16_t* __restrict__ A, const bf16_t* __restrict__ Wt,
                 const float* __restrict__ bias, void* __restrict__ Out,
                 int M, int N, int K) {
  const int wave = threadIdx.x >> 5;
  const unsigned lane = threadIdx.x & 31u;
  const unsigned col = lane & 15u, hi = lane >> 4;
  const int m0 = blockIdx.y * 64 + (wave & 3) * 16;
  const int n0 = blockIdx.x * 64 + (wave >> 2) * 32;
  v8f c0 = {}, c1 = {};
  const bf16_t* a_ptr  = A + (size_t)m0 * K;
  const bf16_t* b0_ptr = Wt + (size_t)n0 * K;
  const bf16_t* b1_ptr = Wt + (size_t)(n0 + 16) * K;
  for (int k0 = 0; k0 < K; k0 += 32) {
    v16bf a  = load_a_frag(a_ptr + k0, K);
    v16bf b0 = load_b_frag(b0_ptr + k0, K);
    v16bf b1 = load_b_frag(b1_ptr + k0, K);
    c0 = wmma_bf16(a, b0, c0);
    c1 = wmma_bf16(a, b1, c1);
  }
  const float bias0 = bias[n0 + col];
  const float bias1 = bias[n0 + 16 + col];
#pragma unroll
  for (int r = 0; r < 8; ++r) {
    const int m = m0 + (int)hi * 8 + r;
    float x0 = c0[r] + bias0;
    float x1 = c1[r] + bias1;
    if constexpr (EPI == EPI_GELU_BF16) {
      x0 = 0.5f * x0 * (1.0f + erff(x0 * 0.70710678118f));
      x1 = 0.5f * x1 * (1.0f + erff(x1 * 0.70710678118f));
    }
    if constexpr (EPI == EPI_F32) {
      float* o = (float*)Out;
      o[(size_t)m * N + n0 + col]      = x0;
      o[(size_t)m * N + n0 + 16 + col] = x1;
    } else if constexpr (EPI == EPI_VT_BF16) {
      // Gemm row m = b*P + j ; col n = h*64 + d ; store V^T as [b,h,d][P]
      bf16_t* o = (bf16_t*)Out;
      const int bb = m >> 12, j = m & (Pc - 1);
      const int na = n0 + col, nbx = n0 + 16 + col;
      o[((size_t)(bb * Hc + (na >> 6))  * HDc + (na & 63))  * Pc + j] = (bf16_t)x0;
      o[((size_t)(bb * Hc + (nbx >> 6)) * HDc + (nbx & 63)) * Pc + j] = (bf16_t)x1;
    } else {
      bf16_t* o = (bf16_t*)Out;
      o[(size_t)m * N + n0 + col]      = (bf16_t)x0;
      o[(size_t)m * N + n0 + 16 + col] = (bf16_t)x1;
    }
  }
}

// ---------------------------------------------------------------------------
// Attention phase 1: logits = base + leaky(base*s_h + ep_b), base = (q.k)/8.
// One block per (b,h,16-query tile); wave w owns keys [w*512, w*512+512).
// Writes logits into d_out attn region; produces per-row max & 1/sum(exp).
// ---------------------------------------------------------------------------
__global__ __launch_bounds__(256)
void attn_logits_kernel(const bf16_t* __restrict__ q, const bf16_t* __restrict__ kk,
                        const float* __restrict__ s_head, const float* __restrict__ epb_p,
                        float* __restrict__ attn, float* __restrict__ rowmax,
                        float* __restrict__ rowinv) {
  const int it = blockIdx.x & 31;          // S/16 = 32 query tiles
  const int h  = (blockIdx.x >> 5) & 7;
  const int bb = blockIdx.x >> 8;
  const int wave = threadIdx.x >> 5;
  const unsigned lane = threadIdx.x & 31u;
  const unsigned col = lane & 15u, hi = lane >> 4;
  const int i0 = it * 16;

  const bf16_t* qbase = q + ((size_t)(bb * Sc + i0)) * TDc + h * HDc;
  v16bf a0 = load_a_frag(qbase, TDc);        // K(d) = 0..31
  v16bf a1 = load_a_frag(qbase + 32, TDc);   // K(d) = 32..63
  const float sh  = s_head[h];
  const float epb = epb_p[0];
  float* attn_bh = attn + ((size_t)(bb * Hc + h) * Sc + i0) * Pc;

  float rmax[8];
#pragma unroll
  for (int r = 0; r < 8; ++r) rmax[r] = -3.0e38f;

  for (int jt = 0; jt < 32; ++jt) {
    const int j0 = wave * 512 + jt * 16;
    const bf16_t* kbase = kk + ((size_t)(bb * Pc + j0)) * TDc + h * HDc;
    v16bf b0 = load_b_frag(kbase, TDc);
    v16bf b1 = load_b_frag(kbase + 32, TDc);
    v8f c = {};
    c = wmma_bf16(a0, b0, c);
    c = wmma_bf16(a1, b1, c);
#pragma unroll
    for (int r = 0; r < 8; ++r) {
      float base_v = c[r] * 0.125f;                 // HD^-0.5
      float e = base_v * sh + epb;
      e = (e > 0.f) ? e : 0.2f * e;                 // leaky_relu(0.2)
      float l = base_v + e;
      attn_bh[(size_t)(hi * 8 + r) * Pc + j0 + col] = l;
      rmax[r] = fmaxf(rmax[r], l);
    }
  }
  // reduce max across the 16 lanes sharing each row group
#pragma unroll
  for (int r = 0; r < 8; ++r)
    for (int off = 1; off < 16; off <<= 1)
      rmax[r] = fmaxf(rmax[r], __shfl_xor(rmax[r], off, 32));

  __shared__ float red[8][16];
  __shared__ float fmax_s[16];
  if (col == 0) {
#pragma unroll
    for (int r = 0; r < 8; ++r) red[wave][hi * 8 + r] = rmax[r];
  }
  __syncthreads();
  if (threadIdx.x < 16) {
    float m = -3.0e38f;
    for (int w = 0; w < 8; ++w) m = fmaxf(m, red[w][threadIdx.x]);
    fmax_s[threadIdx.x] = m;
  }
  __syncthreads();

  float fm[8];
#pragma unroll
  for (int r = 0; r < 8; ++r) fm[r] = fmax_s[hi * 8 + r];
  float rsum[8] = {0, 0, 0, 0, 0, 0, 0, 0};
  // second sweep: re-read own logits (same-wave store->load is ordered; L2-hot)
  for (int jt = 0; jt < 32; ++jt) {
    const int j0 = wave * 512 + jt * 16;
#pragma unroll
    for (int r = 0; r < 8; ++r) {
      float l = attn_bh[(size_t)(hi * 8 + r) * Pc + j0 + col];
      rsum[r] += expf(l - fm[r]);
    }
  }
#pragma unroll
  for (int r = 0; r < 8; ++r)
    for (int off = 1; off < 16; off <<= 1)
      rsum[r] += __shfl_xor(rsum[r], off, 32);
  if (col == 0) {
#pragma unroll
    for (int r = 0; r < 8; ++r) red[wave][hi * 8 + r] = rsum[r];
  }
  __syncthreads();
  if (threadIdx.x < 16) {
    float ssum = 0.f;
    for (int w = 0; w < 8; ++w) ssum += red[w][threadIdx.x];
    const size_t gr = (size_t)(bb * Hc + h) * Sc + i0 + threadIdx.x;
    rowmax[gr] = fmax_s[threadIdx.x];
    rowinv[gr] = 1.0f / ssum;
  }
}

// ---------------------------------------------------------------------------
// Attention phase 2: normalize logits -> probabilities (in place, final attn
// output), convert to bf16 A-fragments, ctx = P @ V^T via WMMA.
// ---------------------------------------------------------------------------
__global__ __launch_bounds__(256)
void attn_pv_kernel(float* __restrict__ attn, const bf16_t* __restrict__ vt,
                    const float* __restrict__ rowmax, const float* __restrict__ rowinv,
                    bf16_t* __restrict__ ctx) {
  const int it = blockIdx.x & 31;
  const int h  = (blockIdx.x >> 5) & 7;
  const int bb = blockIdx.x >> 8;
  const int wave = threadIdx.x >> 5;
  const unsigned lane = threadIdx.x & 31u;
  const unsigned col = lane & 15u, hi = lane >> 4;
  const int i0 = it * 16;

  __shared__ float fmax_s[16], finv_s[16];
  __shared__ float outacc[16 * 64];
  if (threadIdx.x < 16) {
    const size_t gr = (size_t)(bb * Hc + h) * Sc + i0 + threadIdx.x;
    fmax_s[threadIdx.x] = rowmax[gr];
    finv_s[threadIdx.x] = rowinv[gr];
  }
  for (int idx = threadIdx.x; idx < 16 * 64; idx += 256) outacc[idx] = 0.f;
  __syncthreads();

  const unsigned row = col;  // A-fragment row for this lane
  const float fm = fmax_s[row];
  const float fi = finv_s[row];
  float* prow = attn + ((size_t)(bb * Hc + h) * Sc + i0 + row) * Pc;
  const bf16_t* vbase = vt + (size_t)(bb * Hc + h) * HDc * Pc;

  v8f acc[4] = {};
  for (int js = 0; js < 16; ++js) {
    const int j0 = wave * 512 + js * 32;
    const int c0b = j0 + (int)hi * 8;
    v16bf a;
#pragma unroll
    for (int t = 0; t < 8; ++t) {
      float p = expf(prow[c0b + t] - fm) * fi;
      prow[c0b + t] = p;                 // final attn output (fp32)
      a[t] = (bf16_t)p;
    }
#pragma unroll
    for (int t = 0; t < 8; ++t) {
      float p = expf(prow[c0b + 16 + t] - fm) * fi;
      prow[c0b + 16 + t] = p;
      a[8 + t] = (bf16_t)p;
    }
#pragma unroll
    for (int dt = 0; dt < 4; ++dt) {
      v16bf bfrag = load_b_frag(vbase + (size_t)(dt * 16) * Pc + j0, Pc);
      acc[dt] = wmma_bf16(a, bfrag, acc[dt]);
    }
  }
#pragma unroll
  for (int dt = 0; dt < 4; ++dt)
#pragma unroll
    for (int r = 0; r < 8; ++r)
      atomicAdd(&outacc[((int)hi * 8 + r) * 64 + dt * 16 + (int)col], acc[dt][r]);
  __syncthreads();
  for (int idx = threadIdx.x; idx < 16 * 64; idx += 256) {
    const int il = idx >> 6, d = idx & 63;
    ctx[((size_t)(bb * Sc + i0 + il)) * TDc + h * HDc + d] = (bf16_t)outacc[idx];
  }
}

// ---------------------------------------------------------------------------
// Host-side orchestration
// ---------------------------------------------------------------------------
extern "C" void kernel_launch(void* const* d_in, const int* in_sizes, int n_in,
                              void* d_out, int out_size, void* d_ws, size_t ws_size,
                              hipStream_t stream) {
  (void)in_sizes; (void)n_in; (void)out_size; (void)ws_size;
  const float* text  = (const float*)d_in[0];
  const float* shape = (const float*)d_in[1];
  const float* tn_g  = (const float*)d_in[2];
  const float* tn_b  = (const float*)d_in[3];
  const float* sn_g  = (const float*)d_in[4];
  const float* sn_b  = (const float*)d_in[5];
  const float* Wq = (const float*)d_in[6];
  const float* bq = (const float*)d_in[7];
  const float* Wk = (const float*)d_in[8];
  const float* bk = (const float*)d_in[9];
  const float* Wv = (const float*)d_in[10];
  const float* bv = (const float*)d_in[11];
  const float* edge_emb = (const float*)d_in[12];
  const float* ep_w = (const float*)d_in[13];
  const float* ep_b = (const float*)d_in[14];
  const float* Wo = (const float*)d_in[15];
  const float* bo = (const float*)d_in[16];
  const float* on_g = (const float*)d_in[17];
  const float* on_b = (const float*)d_in[18];
  const float* W1 = (const float*)d_in[19];
  const float* b1 = (const float*)d_in[20];
  const float* W2 = (const float*)d_in[21];
  const float* b2 = (const float*)d_in[22];
  const float* fn_g = (const float*)d_in[23];
  const float* fn_b = (const float*)d_in[24];

  float* out_final = (float*)d_out;
  float* attn_out  = out_final + (size_t)Bc * Sc * TDc;  // [B,H,S,P]

  // --- workspace carve (256B aligned) ---
  char* wp = (char*)d_ws;
  auto carve = [&](size_t bytes) -> void* {
    void* p = (void*)wp;
    wp += (bytes + 255) & ~(size_t)255;
    return p;
  };
  bf16_t* tn    = (bf16_t*)carve((size_t)Bc * Sc * TDc * 2);
  bf16_t* sn    = (bf16_t*)carve((size_t)Bc * Pc * SDc * 2);
  bf16_t* WqT   = (bf16_t*)carve((size_t)TDc * TDc * 2);
  bf16_t* WkT   = (bf16_t*)carve((size_t)SDc * TDc * 2);
  bf16_t* WvT   = (bf16_t*)carve((size_t)SDc * TDc * 2);
  bf16_t* WoT   = (bf16_t*)carve((size_t)TDc * TDc * 2);
  bf16_t* W1T   = (bf16_t*)carve((size_t)TDc * FFc * 2);
  bf16_t* W2T   = (bf16_t*)carve((size_t)FFc * TDc * 2);
  bf16_t* qb    = (bf16_t*)carve((size_t)Bc * Sc * TDc * 2);
  bf16_t* kb    = (bf16_t*)carve((size_t)Bc * Pc * TDc * 2);
  bf16_t* vtb   = (bf16_t*)carve((size_t)Bc * Pc * TDc * 2);  // [b,h,d][P]
  bf16_t* ctx   = (bf16_t*)carve((size_t)Bc * Sc * TDc * 2);
  bf16_t* out1b = (bf16_t*)carve((size_t)Bc * Sc * TDc * 2);
  bf16_t* ffh   = (bf16_t*)carve((size_t)Bc * Sc * FFc * 2);
  float*  preo  = (float*)carve((size_t)Bc * Sc * TDc * 4);
  float*  out1f = (float*)carve((size_t)Bc * Sc * TDc * 4);
  float*  ffn2f = (float*)carve((size_t)Bc * Sc * TDc * 4);
  float*  rmax  = (float*)carve((size_t)Bc * Hc * Sc * 4);
  float*  rinv  = (float*)carve((size_t)Bc * Hc * Sc * 4);
  float*  shead = (float*)carve((size_t)Hc * 4);

  // 1. input layernorms -> bf16
  ln_bf16_kernel<<<Bc * Sc, 128, 0, stream>>>(text, tn_g, tn_b, tn, TDc);
  ln_bf16_kernel<<<Bc * Pc, 128, 0, stream>>>(shape, sn_g, sn_b, sn, SDc);

  // 2. weight transpose+convert (tiny; L2-resident afterwards)
  auto tpose = [&](const float* W_, bf16_t* Wt_, int K_, int N_) {
    int total = K_ * N_;
    transpose_bf16_kernel<<<(total + 255) / 256, 256, 0, stream>>>(W_, Wt_, K_, N_);
  };
  tpose(Wq, WqT, TDc, TDc);
  tpose(Wk, WkT, SDc, TDc);
  tpose(Wv, WvT, SDc, TDc);
  tpose(Wo, WoT, TDc, TDc);
  tpose(W1, W1T, TDc, FFc);
  tpose(W2, W2T, FFc, TDc);
  edge_scalar_kernel<<<1, 256, 0, stream>>>(edge_emb, ep_w, shead);

  // 3. Q/K/V projections (V stored transposed per head for the PV GEMM)
  gemm_kernel<EPI_BF16><<<dim3(TDc / 64, (Bc * Sc) / 64), 256, 0, stream>>>(
      tn, WqT, bq, qb, Bc * Sc, TDc, TDc);
  gemm_kernel<EPI_BF16><<<dim3(TDc / 64, (Bc * Pc) / 64), 256, 0, stream>>>(
      sn, WkT, bk, kb, Bc * Pc, TDc, SDc);
  gemm_kernel<EPI_VT_BF16><<<dim3(TDc / 64, (Bc * Pc) / 64), 256, 0, stream>>>(
      sn, WvT, bv, vtb, Bc * Pc, TDc, SDc);

  // 4. attention: logits+softmax stats, then normalize + P@V^T
  attn_logits_kernel<<<Bc * Hc * (Sc / 16), 256, 0, stream>>>(
      qb, kb, shead, ep_b, attn_out, rmax, rinv);
  attn_pv_kernel<<<Bc * Hc * (Sc / 16), 256, 0, stream>>>(
      attn_out, vtb, rmax, rinv, ctx);

  // 5. output projection + residual + LN
  gemm_kernel<EPI_F32><<<dim3(TDc / 64, (Bc * Sc) / 64), 256, 0, stream>>>(
      ctx, WoT, bo, preo, Bc * Sc, TDc, TDc);
  add_ln_kernel<<<Bc * Sc, 128, 0, stream>>>(text, preo, on_g, on_b, out1f, out1b, TDc);

  // 6. FFN (GELU-erf) + residual + final LN -> d_out
  gemm_kernel<EPI_GELU_BF16><<<dim3(FFc / 64, (Bc * Sc) / 64), 256, 0, stream>>>(
      out1b, W1T, b1, ffh, Bc * Sc, FFc, TDc);
  gemm_kernel<EPI_F32><<<dim3(TDc / 64, (Bc * Sc) / 64), 256, 0, stream>>>(
      ffh, W2T, b2, ffn2f, Bc * Sc, TDc, FFc);
  add_ln_kernel<<<Bc * Sc, 128, 0, stream>>>(out1f, ffn2f, fn_g, fn_b, out_final, nullptr, TDc);
}
